// PNAGnn_meta_26027501814368
// MI455X (gfx1250) — matrix operand order repaired and checked by reading, not compiled, not run on hardware
//
#include <hip/hip_runtime.h>
#include <math.h>

// ---------------- problem constants ----------------
constexpr int kV = 32768;
constexpr int kE = 524288;
constexpr int kH = 128;
constexpr int kB = 128;
constexpr int kN = 256;

// ---------------- vector types for WMMA ----------------
typedef __attribute__((ext_vector_type(16))) __bf16 v16bf;
typedef __attribute__((ext_vector_type(8)))  __bf16 v8bf;
typedef __attribute__((ext_vector_type(8)))  float  v8f;

union BF16x16 { v16bf v; v8bf h[2]; };

static __device__ __forceinline__ v8f wmma_bf16(v16bf a, v16bf b, v8f c) {
  // (neg_a, A, neg_b, B, c_mod, C, reuse_a, reuse_b)
  return __builtin_amdgcn_wmma_f32_16x16x32_bf16(false, a, false, b, (short)0, c, false, false);
}

static __device__ __forceinline__ unsigned short f2bf(float f) {
  unsigned u = __float_as_uint(f);
  unsigned r = (u + 0x7FFFu + ((u >> 16) & 1u)) >> 16;  // round-to-nearest-even
  return (unsigned short)r;
}

// ---------------- atomic helpers ----------------
static __device__ __forceinline__ void atAddF(float* p, float v) {
  __hip_atomic_fetch_add(p, v, __ATOMIC_RELAXED, __HIP_MEMORY_SCOPE_AGENT);
}
static __device__ __forceinline__ void atMaxF(float* p, float v) {
  if (v >= 0.0f) atomicMax((int*)p, __float_as_int(v));
  else           atomicMin((unsigned int*)p, __float_as_uint(v));
}
static __device__ __forceinline__ void atMinF(float* p, float v) {
  if (v >= 0.0f) atomicMin((int*)p, __float_as_int(v));
  else           atomicMax((unsigned int*)p, __float_as_uint(v));
}

// ---------------- small utility kernels ----------------
__global__ void zero_f(float* p, int n) {
  int i = blockIdx.x * blockDim.x + threadIdx.x;
  if (i < n) p[i] = 0.0f;
}

__global__ void init_agg(float* sum, float* sumsq, unsigned* mx, unsigned* mn, int n) {
  int i = blockIdx.x * blockDim.x + threadIdx.x;
  if (i < n) {
    sum[i] = 0.0f; sumsq[i] = 0.0f;
    mx[i] = 0xFF800000u;  // -inf
    mn[i] = 0x7F800000u;  // +inf
  }
}

// transpose-convert weights: src [K,N] f32 row-major -> dst [N][K] bf16
__global__ void wt_bf16(const float* __restrict__ src, unsigned short* __restrict__ dst,
                        int K, int N) {
  int i = blockIdx.x * blockDim.x + threadIdx.x;
  if (i >= K * N) return;
  int k = i / N, n = i - k * N;
  dst[n * K + k] = f2bf(src[k * N + n]);
}

__global__ void deg_kernel(const int* __restrict__ dstI, float* __restrict__ deg) {
  int e = blockIdx.x * blockDim.x + threadIdx.x;
  if (e < kE) atAddF(&deg[dstI[e]], 1.0f);
}

__global__ void sumlog_kernel(const float* __restrict__ deg, float* __restrict__ sumlog) {
  __shared__ float red[256];
  int v = blockIdx.x * 256 + threadIdx.x;
  red[threadIdx.x] = (v < kV) ? logf(deg[v] + 1.0f) : 0.0f;
  __syncthreads();
  for (int s = 128; s > 0; s >>= 1) {
    if (threadIdx.x < s) red[threadIdx.x] += red[threadIdx.x + s];
    __syncthreads();
  }
  if (threadIdx.x == 0) atAddF(sumlog, red[0]);
}

__global__ void ampatt_kernel(const float* __restrict__ deg, const float* __restrict__ sumlog,
                              float* __restrict__ amp, float* __restrict__ att) {
  int v = blockIdx.x * blockDim.x + threadIdx.x;
  if (v >= kV) return;
  float avg = sumlog[0] * (1.0f / (float)kV);
  float logd = logf(fmaxf(deg[v], 1.0f) + 1.0f);
  amp[v] = logd / avg;
  att[v] = avg / logd;
}

// ---------------- layer 1 (F=2) scalar kernels ----------------
__global__ void edge1_kernel(const float* __restrict__ x, const int* __restrict__ srcI,
                             const int* __restrict__ dstI,
                             const float* __restrict__ pre1w, const float* __restrict__ pre1b,
                             float* sum1, float* sumsq1, float* mx1, float* mn1) {
  int e = blockIdx.x * blockDim.x + threadIdx.x;
  if (e >= kE) return;
  int dv = dstI[e], sv = srcI[e];
  float c0 = x[dv * 2], c1 = x[dv * 2 + 1], c2 = x[sv * 2], c3 = x[sv * 2 + 1];
  for (int j = 0; j < 2; ++j) {
    float h = c0 * pre1w[j] + c1 * pre1w[2 + j] + c2 * pre1w[4 + j] + c3 * pre1w[6 + j] + pre1b[j];
    int a = dv * 2 + j;
    atAddF(&sum1[a], h);
    atAddF(&sumsq1[a], h * h);
    atMaxF(&mx1[a], h);
    atMinF(&mn1[a], h);
  }
}

__global__ void post1_kernel(const float* __restrict__ x,
                             const float* __restrict__ sum1, const float* __restrict__ sumsq1,
                             const float* __restrict__ mx1f, const float* __restrict__ mn1f,
                             const float* __restrict__ deg,
                             const float* __restrict__ amp, const float* __restrict__ att,
                             const float* __restrict__ post1w, const float* __restrict__ post1b,
                             float* __restrict__ postF, unsigned short* __restrict__ postB) {
  int i = blockIdx.x * blockDim.x + threadIdx.x;
  if (i >= kV * kH) return;
  int v = i >> 7, n = i & 127;
  float d = deg[v], dc = fmaxf(d, 1.0f);
  float in[26];
  in[0] = x[v * 2]; in[1] = x[v * 2 + 1];
  for (int j = 0; j < 2; ++j) {
    float mean = sum1[v * 2 + j] / dc;
    float sd = sqrtf(fmaxf(sumsq1[v * 2 + j] / dc - mean * mean, 0.0f) + 1e-5f);
    in[2 + j] = mean;
    in[4 + j] = sd;
    in[6 + j] = (d > 0.0f) ? mx1f[v * 2 + j] : 0.0f;
    in[8 + j] = (d > 0.0f) ? mn1f[v * 2 + j] : 0.0f;
  }
  float a = amp[v], t = att[v];
  for (int j = 0; j < 8; ++j) { in[10 + j] = in[2 + j] * a; in[18 + j] = in[2 + j] * t; }
  float o = post1b[n];
  for (int k = 0; k < 26; ++k) o += in[k] * post1w[k * kH + n];
  postF[i] = o;
  postB[i] = f2bf(o);
}

// ---------------- WMMA edge message kernel (F=128) ----------------
// wave: 16 edges x 128 outputs; h = cat(state[dst],state[src]) @ pre_w + pre_b,
// aggregated into sum/sumsq/max/min via atomics.
__global__ __launch_bounds__(256) void edge_msg_wmma(
    const unsigned short* __restrict__ stateB,   // [V,128] bf16
    const unsigned short* __restrict__ WtPre,    // [128][256] bf16 (transposed pre_w)
    const float* __restrict__ preB,              // [128]
    const int* __restrict__ srcIdx, const int* __restrict__ dstIdx,
    float* __restrict__ sum, float* __restrict__ sumsq,
    float* __restrict__ mxb, float* __restrict__ mnb) {
  __shared__ unsigned short sW[128 * 256];  // 64 KB
  {
    const uint4* g = (const uint4*)WtPre;
    uint4* s = (uint4*)sW;
    for (int i = threadIdx.x; i < 128 * 256 / 8; i += 256) s[i] = g[i];
  }
  __syncthreads();

  const int wave = threadIdx.x >> 5;
  const int lane = threadIdx.x & 31;
  const int laneHi = lane >> 4;
  const int m16 = lane & 15;
  const int eBase = (blockIdx.x * 8 + wave) * 16;

  const int nd = dstIdx[eBase + m16];
  const int ns = srcIdx[eBase + m16];

  v8f acc[8] = {};

  for (int ks = 0; ks < 8; ++ks) {
    const int node = (ks < 4) ? nd : ns;
    const int col = (ks & 3) * 32 + laneHi * 16;
    const unsigned short* rp = stateB + (size_t)node * kH + col;
    BF16x16 a;
    a.h[0] = *(const v8bf*)(rp);
    a.h[1] = *(const v8bf*)(rp + 8);
#pragma unroll
    for (int t = 0; t < 8; ++t) {
      const int n = t * 16 + m16;
      v16bf b = *(const v16bf*)(sW + n * 256 + ks * 32 + laneHi * 16);
      acc[t] = wmma_bf16(a.v, b, acc[t]);
    }
  }

#pragma unroll
  for (int t = 0; t < 8; ++t) {
    const int n = t * 16 + m16;
    const float bias = preB[n];
#pragma unroll
    for (int r = 0; r < 8; ++r) {
      const int m = laneHi * 8 + r;
      const int node = __shfl(nd, m, 32);
      const float h = acc[t][r] + bias;
      const size_t addr = (size_t)node * kH + n;
      atAddF(&sum[addr], h);
      atAddF(&sumsq[addr], h * h);
      atMaxF(&mxb[addr], h);
      atMinF(&mnb[addr], h);
    }
  }
}

// finalize aggregations -> agg bf16 [V,512] = [mean,std,max,min]
__global__ void finalize_agg(const float* __restrict__ sum, const float* __restrict__ sumsq,
                             const float* __restrict__ mxf, const float* __restrict__ mnf,
                             const float* __restrict__ deg, unsigned short* __restrict__ aggB) {
  int i = blockIdx.x * blockDim.x + threadIdx.x;
  if (i >= kV * kH) return;
  int v = i >> 7, f = i & 127;
  float d = deg[v], dc = fmaxf(d, 1.0f);
  float mean = sum[i] / dc;
  float sd = sqrtf(fmaxf(sumsq[i] / dc - mean * mean, 0.0f) + 1e-5f);
  float mx = (d > 0.0f) ? mxf[i] : 0.0f;
  float mn = (d > 0.0f) ? mnf[i] : 0.0f;
  unsigned short* row = aggB + (size_t)v * 512;
  row[f] = f2bf(mean);
  row[128 + f] = f2bf(sd);
  row[256 + f] = f2bf(mx);
  row[384 + f] = f2bf(mn);
}

// ---------------- WMMA post kernel ----------------
// out = x@Wx + agg@W1 + amp*(agg@W2) + att*(agg@W3) + post_b
// WtPost: [128][1664] bf16, k layout: [0,128)=x, [128,640)=W1, [640,1152)=W2, [1152,1664)=W3
__global__ __launch_bounds__(256) void node_post_wmma(
    const unsigned short* __restrict__ stateB, const unsigned short* __restrict__ aggB,
    const unsigned short* __restrict__ WtPost, const float* __restrict__ postBias,
    const float* __restrict__ amp, const float* __restrict__ att,
    float* __restrict__ postF, unsigned short* __restrict__ postB) {
  const int wave = threadIdx.x >> 5;
  const int lane = threadIdx.x & 31;
  const int laneHi = lane >> 4;
  const int lid = lane & 15;
  const int gw = blockIdx.x * 8 + wave;   // (V/16)*8 waves
  const int vTile = gw >> 3;
  const int nTile = gw & 7;
  const int vBase = vTile * 16;
  const int n = nTile * 16 + lid;

  v8f a0 = {}, a1 = {}, a2 = {}, a3 = {};
  const unsigned short* wrow = WtPost + (size_t)n * 1664;

  const unsigned short* xp = stateB + (size_t)(vBase + lid) * kH;
#pragma unroll
  for (int ks = 0; ks < 4; ++ks) {
    BF16x16 a;
    const int c = ks * 32 + laneHi * 16;
    a.h[0] = *(const v8bf*)(xp + c);
    a.h[1] = *(const v8bf*)(xp + c + 8);
    v16bf b = *(const v16bf*)(wrow + ks * 32 + laneHi * 16);
    a0 = wmma_bf16(a.v, b, a0);
  }
  const unsigned short* gp = aggB + (size_t)(vBase + lid) * 512;
#pragma unroll
  for (int ks = 0; ks < 16; ++ks) {
    BF16x16 a;
    const int c = ks * 32 + laneHi * 16;
    a.h[0] = *(const v8bf*)(gp + c);
    a.h[1] = *(const v8bf*)(gp + c + 8);
    const int kb = ks * 32 + laneHi * 16;
    v16bf b1 = *(const v16bf*)(wrow + 128 + kb);
    v16bf b2 = *(const v16bf*)(wrow + 640 + kb);
    v16bf b3 = *(const v16bf*)(wrow + 1152 + kb);
    a1 = wmma_bf16(a.v, b1, a1);
    a2 = wmma_bf16(a.v, b2, a2);
    a3 = wmma_bf16(a.v, b3, a3);
  }

  const float bias = postBias[n];
#pragma unroll
  for (int r = 0; r < 8; ++r) {
    const int m = laneHi * 8 + r;
    const int v = vBase + m;
    const float o = a0[r] + a1[r] + amp[v] * a2[r] + att[v] * a3[r] + bias;
    postF[(size_t)v * kH + n] = o;
    postB[(size_t)v * kH + n] = f2bf(o);
  }
}

// ---------------- generic [V,128]x[128,128] WMMA GEMM (+bias, optional leaky) -------
__global__ __launch_bounds__(256) void lin_wmma(
    const unsigned short* __restrict__ inB, const unsigned short* __restrict__ Wt,
    const float* __restrict__ bias, float* __restrict__ outF,
    unsigned short* __restrict__ outB, int applyLeaky) {
  const int wave = threadIdx.x >> 5;
  const int lane = threadIdx.x & 31;
  const int laneHi = lane >> 4;
  const int lid = lane & 15;
  const int gw = blockIdx.x * 8 + wave;
  const int vTile = gw >> 3;
  const int nTile = gw & 7;
  const int vBase = vTile * 16;
  const int n = nTile * 16 + lid;

  v8f acc = {};
  const unsigned short* xp = inB + (size_t)(vBase + lid) * kH;
  const unsigned short* wrow = Wt + (size_t)n * kH;
#pragma unroll
  for (int ks = 0; ks < 4; ++ks) {
    BF16x16 a;
    const int c = ks * 32 + laneHi * 16;
    a.h[0] = *(const v8bf*)(xp + c);
    a.h[1] = *(const v8bf*)(xp + c + 8);
    v16bf b = *(const v16bf*)(wrow + ks * 32 + laneHi * 16);
    acc = wmma_bf16(a.v, b, acc);
  }
  const float bv = bias[n];
#pragma unroll
  for (int r = 0; r < 8; ++r) {
    const int m = laneHi * 8 + r;
    const int v = vBase + m;
    float o = acc[r] + bv;
    if (applyLeaky) o = (o > 0.0f) ? o : 0.01f * o;
    outF[(size_t)v * kH + n] = o;
    if (outB) outB[(size_t)v * kH + n] = f2bf(o);
  }
}

// ---------------- readout tail: 128 -> 8 -> 3 + per-batch sq-err ----------------
__global__ void readout_tail(const float* __restrict__ h,
                             const float* __restrict__ r2w, const float* __restrict__ r2b,
                             const float* __restrict__ r3w, const float* __restrict__ r3b,
                             const float* __restrict__ tgt, float* __restrict__ out,
                             float* __restrict__ lossb) {
  int v = blockIdx.x * blockDim.x + threadIdx.x;
  if (v >= kV) return;
  const float* hr = h + (size_t)v * kH;
  float mid[8];
  for (int j = 0; j < 8; ++j) mid[j] = r2b[j];
  for (int k = 0; k < kH; ++k) {
    float hv = hr[k];
    for (int j = 0; j < 8; ++j) mid[j] += hv * r2w[k * 8 + j];
  }
  for (int j = 0; j < 8; ++j) { float m = mid[j]; mid[j] = (m > 0.0f) ? m : 0.01f * m; }
  float s = 0.0f;
  for (int c = 0; c < 3; ++c) {
    float o = r3b[c];
    for (int j = 0; j < 8; ++j) o += mid[j] * r3w[j * 3 + c];
    o = (o > 0.0f) ? o : 0.01f * o;
    out[v * 3 + c] = o;
    float d = o - tgt[v * 3 + c];
    s += d * d;
  }
  atAddF(&lossb[v >> 8], s);
}

__global__ void loss_finalize(const float* __restrict__ lossb, float* __restrict__ out) {
  __shared__ float red[128];
  int b = threadIdx.x;
  float pb = lossb[b] * (1.0f / ((float)kN * 3.0f));
  out[kV * 3 + 1 + b] = pb;
  red[b] = pb;
  __syncthreads();
  for (int s = 64; s > 0; s >>= 1) {
    if (b < s) red[b] += red[b + s];
    __syncthreads();
  }
  if (b == 0) out[kV * 3] = red[0] * (1.0f / (float)kB);
}

// ---------------- host orchestration ----------------
extern "C" void kernel_launch(void* const* d_in, const int* in_sizes, int n_in,
                              void* d_out, int out_size, void* d_ws, size_t ws_size,
                              hipStream_t stream) {
  (void)in_sizes; (void)n_in; (void)out_size; (void)ws_size;
  const float* x        = (const float*)d_in[0];
  const int*   ei       = (const int*)d_in[1];
  const float* target_n = (const float*)d_in[2];
  // d_in[3] target_g unused
  const float* pre1_w   = (const float*)d_in[4];
  const float* pre1_b   = (const float*)d_in[5];
  const float* post1_w  = (const float*)d_in[6];
  const float* post1_b  = (const float*)d_in[7];
  const float* lin1_w   = (const float*)d_in[8];
  const float* lin1_b   = (const float*)d_in[9];
  const float* pre_w    = (const float*)d_in[10];
  const float* pre_b    = (const float*)d_in[11];
  const float* post_w   = (const float*)d_in[12];
  const float* post_b   = (const float*)d_in[13];
  const float* lin_w    = (const float*)d_in[14];
  const float* lin_b    = (const float*)d_in[15];
  const float* r1_w     = (const float*)d_in[16];
  const float* r1_b     = (const float*)d_in[17];
  const float* r2_w     = (const float*)d_in[18];
  const float* r2_b     = (const float*)d_in[19];
  const float* r3_w     = (const float*)d_in[20];
  const float* r3_b     = (const float*)d_in[21];
  const int* srcI = ei;
  const int* dstI = ei + kE;
  float* out = (float*)d_out;

  // workspace carve-out
  char* w = (char*)d_ws;
  auto alloc = [&](size_t bytes) -> void* {
    void* p = (void*)w;
    w += (bytes + 255) & ~(size_t)255;
    return p;
  };
  float* deg     = (float*)alloc((size_t)kV * 4);
  float* sumlog  = (float*)alloc(256);
  float* amp     = (float*)alloc((size_t)kV * 4);
  float* att     = (float*)alloc((size_t)kV * 4);
  float* stateF  = (float*)alloc((size_t)kV * kH * 4);
  unsigned short* stateB = (unsigned short*)alloc((size_t)kV * kH * 2);
  float* postF   = (float*)alloc((size_t)kV * kH * 4);
  unsigned short* postB  = (unsigned short*)alloc((size_t)kV * kH * 2);
  float* aSum    = (float*)alloc((size_t)kV * kH * 4);
  float* aSq     = (float*)alloc((size_t)kV * kH * 4);
  float* aMx     = (float*)alloc((size_t)kV * kH * 4);
  float* aMn     = (float*)alloc((size_t)kV * kH * 4);
  unsigned short* aggB = (unsigned short*)alloc((size_t)kV * 512 * 2);
  float* sum1    = (float*)alloc((size_t)kV * 2 * 4);
  float* sumsq1  = (float*)alloc((size_t)kV * 2 * 4);
  float* mx1     = (float*)alloc((size_t)kV * 2 * 4);
  float* mn1     = (float*)alloc((size_t)kV * 2 * 4);
  unsigned short* WtPre  = (unsigned short*)alloc((size_t)128 * 256 * 2);
  unsigned short* WtPost = (unsigned short*)alloc((size_t)128 * 1664 * 2);
  unsigned short* WtLin1 = (unsigned short*)alloc((size_t)128 * 128 * 2);
  unsigned short* WtLin  = (unsigned short*)alloc((size_t)128 * 128 * 2);
  unsigned short* WtR1   = (unsigned short*)alloc((size_t)128 * 128 * 2);
  float* lossb   = (float*)alloc((size_t)kB * 4);

  (void)stateF;
  auto cdiv = [](int a, int b) { return (a + b - 1) / b; };

  // weight transpose/convert to bf16
  wt_bf16<<<cdiv(256 * 128, 256), 256, 0, stream>>>(pre_w, WtPre, 256, 128);
  wt_bf16<<<cdiv(1664 * 128, 256), 256, 0, stream>>>(post_w, WtPost, 1664, 128);
  wt_bf16<<<cdiv(128 * 128, 256), 256, 0, stream>>>(lin1_w, WtLin1, 128, 128);
  wt_bf16<<<cdiv(128 * 128, 256), 256, 0, stream>>>(lin_w, WtLin, 128, 128);
  wt_bf16<<<cdiv(128 * 128, 256), 256, 0, stream>>>(r1_w, WtR1, 128, 128);

  // degree statistics
  zero_f<<<cdiv(kV, 256), 256, 0, stream>>>(deg, kV);
  zero_f<<<1, 256, 0, stream>>>(sumlog, 1);
  deg_kernel<<<cdiv(kE, 256), 256, 0, stream>>>(dstI, deg);
  sumlog_kernel<<<kV / 256, 256, 0, stream>>>(deg, sumlog);
  ampatt_kernel<<<cdiv(kV, 256), 256, 0, stream>>>(deg, sumlog, amp, att);

  // conv layer 1 (F=2)
  init_agg<<<cdiv(kV * 2, 256), 256, 0, stream>>>(sum1, sumsq1, (unsigned*)mx1, (unsigned*)mn1, kV * 2);
  edge1_kernel<<<cdiv(kE, 256), 256, 0, stream>>>(x, srcI, dstI, pre1_w, pre1_b, sum1, sumsq1, mx1, mn1);
  post1_kernel<<<cdiv(kV * kH, 256), 256, 0, stream>>>(x, sum1, sumsq1, mx1, mn1, deg, amp, att,
                                                       post1_w, post1_b, postF, postB);
  lin_wmma<<<(kV / 16) * 8 / 8, 256, 0, stream>>>(postB, WtLin1, lin1_b, stateF, stateB, 0);

  // 3 heavy PNA conv layers (F=128)
  for (int layer = 0; layer < 3; ++layer) {
    init_agg<<<cdiv(kV * kH, 256), 256, 0, stream>>>(aSum, aSq, (unsigned*)aMx, (unsigned*)aMn, kV * kH);
    edge_msg_wmma<<<kE / 16 / 8, 256, 0, stream>>>(stateB, WtPre, pre_b, srcI, dstI, aSum, aSq, aMx, aMn);
    finalize_agg<<<cdiv(kV * kH, 256), 256, 0, stream>>>(aSum, aSq, aMx, aMn, deg, aggB);
    node_post_wmma<<<(kV / 16) * 8 / 8, 256, 0, stream>>>(stateB, aggB, WtPost, post_b, amp, att, postF, postB);
    lin_wmma<<<(kV / 16) * 8 / 8, 256, 0, stream>>>(postB, WtLin, lin_b, stateF, stateB, 0);
  }

  // readout
  lin_wmma<<<(kV / 16) * 8 / 8, 256, 0, stream>>>(stateB, WtR1, r1_b, postF, (unsigned short*)nullptr, 1);
  zero_f<<<1, 256, 0, stream>>>(lossb, kB);
  readout_tail<<<cdiv(kV, 256), 256, 0, stream>>>(postF, r2_w, r2_b, r3_w, r3_b, target_n, out, lossb);
  loss_finalize<<<1, 128, 0, stream>>>(lossb, out);
}
